// GPT2Attention_4054449127551
// MI455X (gfx1250) — matrix operand order repaired and checked by reading, not compiled, not run on hardware
//
#include <hip/hip_runtime.h>
#include <hip/hip_bf16.h>

// ---------------------------------------------------------------------------
// GPT-2 attention (SVD low-rank K/V) for MI455X / gfx1250.
// All matmuls via v_wmma_f32_16x16x32_bf16 (wave32); GEMM tiles staged into
// LDS with double-buffered GLOBAL_LOAD_ASYNC_TO_LDS_B128 when available.
// ---------------------------------------------------------------------------

typedef __attribute__((ext_vector_type(16))) __bf16 v16bf;
typedef __attribute__((ext_vector_type(8)))  __bf16 v8bf;
typedef __attribute__((ext_vector_type(8)))  float  v8f;
typedef __attribute__((ext_vector_type(4)))  int    v4i;

#define AS1 __attribute__((address_space(1)))
#define AS3 __attribute__((address_space(3)))

constexpr int BB   = 4;
constexpr int SS   = 2048;
constexpr int DM   = 1024;     // model dim
constexpr int NH   = 16;       // heads
constexpr int HDIM = 64;       // head dim
constexpr int RR   = 512;      // low rank
constexpr int MTOT = BB * SS;  // 8192 token rows

__device__ __forceinline__ v8f wmma_bf16(v16bf a, v16bf b, v8f c) {
    return __builtin_amdgcn_wmma_f32_16x16x32_bf16(
        false, a, false, b, (short)0, c, false, false);
}

// ---------------------------------------------------------------------------
// Async global->LDS copy (CDNA5 ASYNCcnt path), with sync fallback.
// Builtin signature (from hipcc diagnostic): param0 = AS1 v4i*, so the
// global source comes first; LDS destination is the AS3 pointer.
// ---------------------------------------------------------------------------
#if __has_builtin(__builtin_amdgcn_global_load_async_to_lds_b128) && \
    __has_builtin(__builtin_amdgcn_s_wait_asynccnt)
#define HAS_ASYNC_LDS 1
#else
#define HAS_ASYNC_LDS 0
#endif

__device__ __forceinline__ void copy8_async(const __bf16* g, __bf16* l) {
#if HAS_ASYNC_LDS
    __builtin_amdgcn_global_load_async_to_lds_b128(
        (AS1 v4i*)g, (AS3 v4i*)l, 0, 0);
#else
    *(v8bf*)l = *(const v8bf*)g;
#endif
}
__device__ __forceinline__ void async_join() {
#if HAS_ASYNC_LDS
    __builtin_amdgcn_s_wait_asynccnt(0);
#endif
}

// ---------------------------------------------------------------------------
// fp32 -> bf16 cast (plain)
// ---------------------------------------------------------------------------
__global__ void cast_f32_bf16(const float* __restrict__ src,
                              __bf16* __restrict__ dst, size_t n) {
    size_t i = (size_t)blockIdx.x * blockDim.x + threadIdx.x;
    size_t stride = (size_t)gridDim.x * blockDim.x;
    for (; i < n; i += stride) dst[i] = (__bf16)src[i];
}

// ---------------------------------------------------------------------------
// fp32 [N,K] -> bf16 [K,N] cast + transpose (32x32 tiles, 256 threads)
// ---------------------------------------------------------------------------
__global__ __launch_bounds__(256)
void cast_transpose(const float* __restrict__ src, __bf16* __restrict__ dst,
                    int N, int K) {
    __shared__ float tile[32][33];
    const int n0 = blockIdx.x * 32;
    const int k0 = blockIdx.y * 32;
    const int tx = threadIdx.x & 31;
    const int ty = threadIdx.x >> 5;   // 0..7
#pragma unroll
    for (int i = ty; i < 32; i += 8)
        tile[i][tx] = src[(size_t)(n0 + i) * K + k0 + tx];
    __syncthreads();
#pragma unroll
    for (int i = ty; i < 32; i += 8)
        dst[(size_t)(k0 + i) * N + n0 + tx] = (__bf16)tile[tx][i];
}

// ---------------------------------------------------------------------------
// Fragment loaders (contiguous k order: lane half g holds k = g*16..g*16+15)
// ---------------------------------------------------------------------------
// A fragment 16x32 (MxK), row-major, row stride ldk.
__device__ __forceinline__ v16bf load_a_frag(const __bf16* __restrict__ base,
                                             int ldk, int lane) {
    const int g = lane >> 4, m = lane & 15;
    const __bf16* row = base + m * ldk + g * 16;
    v8bf lo = *(const v8bf*)(row);        // k = g*16 ..
    v8bf hi = *(const v8bf*)(row + 8);    // k = g*16+8 ..
    v16bf a;
#pragma unroll
    for (int i = 0; i < 8; ++i) { a[i] = lo[i]; a[i + 8] = hi[i]; }
    return a;
}

// B fragment 32x16 (KxN) from K-major tile with row stride ldn.
__device__ __forceinline__ v16bf load_b_frag(const __bf16* __restrict__ base,
                                             int ldn, int lane) {
    const int g = lane >> 4, n = lane & 15;
    v16bf b;
#pragma unroll
    for (int e = 0; e < 16; ++e) b[e] = base[(e + g * 16) * ldn + n];
    return b;
}

// ---------------------------------------------------------------------------
// bf16 WMMA GEMM: C[M,N] = A[M,K] @ W[K,N] + bias
// Block 256 threads = 8 waves (4 M-waves x 2 N-waves); block tile 128x128,
// wave tile 32x64 (2x4 fragments, 8 WMMA / K-step), K-step 32.
// LDS double-buffered; tiles staged via async global->LDS when available.
// Requires M%128==0, N%128==0, K%32==0 (all shapes here comply).
// ---------------------------------------------------------------------------
template <bool OUT_F32, bool HAS_BIAS>
__global__ __launch_bounds__(256)
void gemm_bf16(const __bf16* __restrict__ A, const __bf16* __restrict__ W,
               const float* __restrict__ bias, void* __restrict__ Cout,
               int M, int N, int K) {
    __shared__ __bf16 As[2][128 * 32];   // [m][k]
    __shared__ __bf16 Bs[2][32 * 128];   // [k][n]

    const int tid  = threadIdx.x;
    const int lane = tid & 31;
    const int wave = tid >> 5;
    const int wm   = wave >> 1;          // 0..3
    const int wn   = wave & 1;           // 0..1
    const int g    = lane >> 4;
    const int l16  = lane & 15;
    const int bm = blockIdx.y * 128;
    const int bn = blockIdx.x * 128;

    // per-thread staging assignment (each thread: 16 elems of A, 16 of B)
    const int ar = tid >> 1, ak = (tid & 1) * 16;       // A: 128 rows x 32 k
    const int bk = tid >> 3, bnn = (tid & 7) * 16;      // B: 32 k x 128 n

    auto stage = [&](int k0, int buf) {
        const __bf16* ga = A + (size_t)(bm + ar) * K + k0 + ak;
        __bf16* la = &As[buf][ar * 32 + ak];
        copy8_async(ga, la);
        copy8_async(ga + 8, la + 8);
        const __bf16* gb = W + (size_t)(k0 + bk) * N + bn + bnn;
        __bf16* lb = &Bs[buf][bk * 128 + bnn];
        copy8_async(gb, lb);
        copy8_async(gb + 8, lb + 8);
    };

    v8f acc[2][4] = {};
    const int nsteps = K >> 5;

    stage(0, 0);
    for (int s = 0; s < nsteps; ++s) {
        async_join();          // own wave's async tile-s writes done
        __syncthreads();       // everyone's writes visible; prev reads done
        if (s + 1 < nsteps) stage((s + 1) * 32, (s + 1) & 1);

        const __bf16* Asb = As[s & 1];
        const __bf16* Bsb = Bs[s & 1];
        v16bf af[2], bfr[4];
#pragma unroll
        for (int fm = 0; fm < 2; ++fm)
            af[fm] = load_a_frag(Asb + (wm * 32 + fm * 16) * 32, 32, lane);
#pragma unroll
        for (int fn = 0; fn < 4; ++fn)
            bfr[fn] = load_b_frag(Bsb + wn * 64 + fn * 16, 128, lane);
#pragma unroll
        for (int fm = 0; fm < 2; ++fm)
#pragma unroll
            for (int fn = 0; fn < 4; ++fn)
                acc[fm][fn] = wmma_bf16(af[fm], bfr[fn], acc[fm][fn]);
    }

    // epilogue: D frag elem r -> row r+g*8, col l16
#pragma unroll
    for (int fm = 0; fm < 2; ++fm) {
#pragma unroll
        for (int fn = 0; fn < 4; ++fn) {
#pragma unroll
            for (int r = 0; r < 8; ++r) {
                int row = bm + wm * 32 + fm * 16 + r + g * 8;
                int col = bn + wn * 64 + fn * 16 + l16;
                float v = acc[fm][fn][r];
                if (HAS_BIAS) v += bias[col];
                if (OUT_F32) {
                    ((float*)Cout)[(size_t)row * N + col] = v;
                } else {
                    ((__bf16*)Cout)[(size_t)row * N + col] = (__bf16)v;
                }
            }
        }
    }
}

// ---------------------------------------------------------------------------
// Flash-style causal attention, head_dim 64, scale 1/8.
// Block 128 threads = 4 waves; wave owns 16 query rows; 32-wide key tiles.
// Q/K/V/O are [B,S,D] bf16 with head h at columns h*64..h*64+63.
// ---------------------------------------------------------------------------
__global__ __launch_bounds__(128)
void attn_flash(const __bf16* __restrict__ Q, const __bf16* __restrict__ Kt,
                const __bf16* __restrict__ Vt, __bf16* __restrict__ O) {
    const int bh   = blockIdx.y;
    const int b    = bh / NH;
    const int h    = bh % NH;
    const int wave = threadIdx.x >> 5;
    const int lane = threadIdx.x & 31;
    const int g    = lane >> 4;
    const int l16  = lane & 15;
    const int qbase = blockIdx.x * 64 + wave * 16;
    const int qlast = qbase + 15;

    const size_t headoff = ((size_t)b * SS) * DM + (size_t)h * HDIM;
    const __bf16* Qb = Q + headoff;
    const __bf16* Kb = Kt + headoff;
    const __bf16* Vb = Vt + headoff;

    v16bf qa[2];
#pragma unroll
    for (int st = 0; st < 2; ++st) {
        const __bf16* row = Qb + (size_t)(qbase + l16) * DM + st * 32 + g * 16;
        v8bf lo = *(const v8bf*)(row);
        v8bf hi = *(const v8bf*)(row + 8);
#pragma unroll
        for (int i = 0; i < 8; ++i) { qa[st][i] = lo[i]; qa[st][i + 8] = hi[i]; }
    }

    __shared__ __bf16 Pshm[4][16 * 32];
    __bf16* P = Pshm[wave];

    v8f o[4] = {};
    float rm[8], rs[8];
#pragma unroll
    for (int r = 0; r < 8; ++r) { rm[r] = -3.0e38f; rs[r] = 0.0f; }

    for (int kt = 0; kt * 32 <= qlast; ++kt) {
        const int kb32 = kt * 32;

        // scores: two 16-key N-fragments, contraction dim 64 (2 WMMA each)
        v8f s[2];
#pragma unroll
        for (int f = 0; f < 2; ++f) {
            const int key = kb32 + f * 16 + l16;
            const __bf16* kp = Kb + (size_t)key * DM + g * 16;
            v16bf b0 = *(const v16bf*)(kp);
            v16bf b1 = *(const v16bf*)(kp + 32);
            v8f z = {};
            s[f] = wmma_bf16(qa[0], b0, z);
            s[f] = wmma_bf16(qa[1], b1, s[f]);
        }

        // scale + causal mask
        float pv[2][8];
#pragma unroll
        for (int f = 0; f < 2; ++f)
#pragma unroll
            for (int r = 0; r < 8; ++r) {
                int row = qbase + r + g * 8;
                int key = kb32 + f * 16 + l16;
                pv[f][r] = (key <= row) ? s[f][r] * 0.125f : -3.0e38f;
            }

        // online softmax (reduce across the 16-lane N group)
#pragma unroll
        for (int r = 0; r < 8; ++r) {
            float t = fmaxf(pv[0][r], pv[1][r]);
#pragma unroll
            for (int off = 1; off < 16; off <<= 1)
                t = fmaxf(t, __shfl_xor(t, off, 32));
            float nm = fmaxf(rm[r], t);
            float c  = __expf(rm[r] - nm);
            rm[r] = nm;
            float p0 = __expf(pv[0][r] - nm);
            float p1 = __expf(pv[1][r] - nm);
            pv[0][r] = p0; pv[1][r] = p1;
            float ps = p0 + p1;
#pragma unroll
            for (int off = 1; off < 16; off <<= 1)
                ps += __shfl_xor(ps, off, 32);
            rs[r] = rs[r] * c + ps;
#pragma unroll
            for (int j = 0; j < 4; ++j) o[j][r] *= c;
        }

        // P: D-frag -> A-frag relayout via per-wave LDS (same-wave RAW)
#pragma unroll
        for (int f = 0; f < 2; ++f)
#pragma unroll
            for (int r = 0; r < 8; ++r)
                P[(r + g * 8) * 32 + f * 16 + l16] = (__bf16)pv[f][r];
        v16bf pa;
        {
            const __bf16* prow = P + l16 * 32 + g * 16;
            v8bf lo = *(const v8bf*)(prow);
            v8bf hi = *(const v8bf*)(prow + 8);
#pragma unroll
            for (int i = 0; i < 8; ++i) { pa[i] = lo[i]; pa[i + 8] = hi[i]; }
        }

        // O += P @ V
#pragma unroll
        for (int fn = 0; fn < 4; ++fn) {
            v16bf vb;
#pragma unroll
            for (int e = 0; e < 16; ++e) {
                int key = kb32 + e + g * 16;
                vb[e] = Vb[(size_t)key * DM + fn * 16 + l16];
            }
            o[fn] = wmma_bf16(pa, vb, o[fn]);
        }
    }

    __bf16* Ob = O + headoff;
#pragma unroll
    for (int fn = 0; fn < 4; ++fn)
#pragma unroll
        for (int r = 0; r < 8; ++r) {
            int row = qbase + r + g * 8;
            Ob[(size_t)row * DM + fn * 16 + l16] = (__bf16)(o[fn][r] / rs[r]);
        }
}

// ---------------------------------------------------------------------------
// Host launcher
// ---------------------------------------------------------------------------
extern "C" void kernel_launch(void* const* d_in, const int* in_sizes, int n_in,
                              void* d_out, int out_size, void* d_ws, size_t ws_size,
                              hipStream_t stream) {
    (void)in_sizes; (void)n_in; (void)out_size; (void)ws_size;

    const float* hs     = (const float*)d_in[0];
    const float* WQ_w   = (const float*)d_in[1];
    const float* WQ_b   = (const float*)d_in[2];
    const float* WK_A_w = (const float*)d_in[3];
    const float* WK_B_w = (const float*)d_in[4];
    const float* WK_B_b = (const float*)d_in[5];
    const float* WV_A_w = (const float*)d_in[6];
    const float* WV_B_w = (const float*)d_in[7];
    const float* WV_B_b = (const float*)d_in[8];
    const float* CP_w   = (const float*)d_in[9];
    const float* CP_b   = (const float*)d_in[10];

    char* ws = (char*)d_ws;
    size_t off = 0;
    auto alloc_bf16 = [&](size_t elems) -> __bf16* {
        __bf16* p = (__bf16*)(ws + off);
        off += ((elems * sizeof(__bf16)) + 255) & ~(size_t)255;
        return p;
    };

    __bf16* hsb  = alloc_bf16((size_t)MTOT * DM);
    __bf16* wqt  = alloc_bf16((size_t)DM * DM);   // [K=1024][N=1024]
    __bf16* wkat = alloc_bf16((size_t)DM * RR);   // [K=1024][N=512]
    __bf16* wkbt = alloc_bf16((size_t)RR * DM);   // [K=512][N=1024]
    __bf16* wvat = alloc_bf16((size_t)DM * RR);
    __bf16* wvbt = alloc_bf16((size_t)RR * DM);
    __bf16* wcpb = alloc_bf16((size_t)DM * DM);   // already [K,N]
    __bf16* Qb   = alloc_bf16((size_t)MTOT * DM);
    __bf16* Kab  = alloc_bf16((size_t)MTOT * RR);
    __bf16* Kfb  = alloc_bf16((size_t)MTOT * DM);
    __bf16* Vab  = alloc_bf16((size_t)MTOT * RR);
    __bf16* Vfb  = alloc_bf16((size_t)MTOT * DM);
    __bf16* Ob   = alloc_bf16((size_t)MTOT * DM);

    auto cast = [&](const float* s, __bf16* d, size_t n) {
        int blocks = (int)((n + 255) / 256);
        if (blocks > 2048) blocks = 2048;
        cast_f32_bf16<<<blocks, 256, 0, stream>>>(s, d, n);
    };
    auto castT = [&](const float* s, __bf16* d, int N, int K) {
        cast_transpose<<<dim3(N / 32, K / 32), 256, 0, stream>>>(s, d, N, K);
    };

    cast (hs,     hsb,  (size_t)MTOT * DM);
    castT(WQ_w,   wqt,  DM, DM);    // [1024,1024] -> [1024,1024]^T
    castT(WK_A_w, wkat, RR, DM);    // [512,1024]  -> [1024,512]
    castT(WK_B_w, wkbt, DM, RR);    // [1024,512]  -> [512,1024]
    castT(WV_A_w, wvat, RR, DM);
    castT(WV_B_w, wvbt, DM, RR);
    cast (CP_w,   wcpb, (size_t)DM * DM);   // Conv1D weight already [in,out]

    // Q = hs @ WQ^T + b
    gemm_bf16<false, true><<<dim3(DM / 128, MTOT / 128), 256, 0, stream>>>(
        hsb, wqt, WQ_b, Qb, MTOT, DM, DM);
    // Ka = hs @ WK_A^T
    gemm_bf16<false, false><<<dim3(RR / 128, MTOT / 128), 256, 0, stream>>>(
        hsb, wkat, nullptr, Kab, MTOT, RR, DM);
    // K = Ka @ WK_B^T + b
    gemm_bf16<false, true><<<dim3(DM / 128, MTOT / 128), 256, 0, stream>>>(
        Kab, wkbt, WK_B_b, Kfb, MTOT, DM, RR);
    // Va = hs @ WV_A^T
    gemm_bf16<false, false><<<dim3(RR / 128, MTOT / 128), 256, 0, stream>>>(
        hsb, wvat, nullptr, Vab, MTOT, RR, DM);
    // V = Va @ WV_B^T + b
    gemm_bf16<false, true><<<dim3(DM / 128, MTOT / 128), 256, 0, stream>>>(
        Vab, wvbt, WV_B_b, Vfb, MTOT, DM, RR);

    // causal flash attention
    attn_flash<<<dim3(SS / 64, BB * NH), 128, 0, stream>>>(Qb, Kfb, Vfb, Ob);

    // out = attn @ c_proj_w + c_proj_b (fp32 out)
    gemm_bf16<true, true><<<dim3(DM / 128, MTOT / 128), 256, 0, stream>>>(
        Ob, wcpb, CP_b, (float*)d_out, MTOT, DM, DM);
}
